// DistanceField_45707041964605
// MI455X (gfx1250) — matrix operand order repaired and checked by reading, not compiled, not run on hardware
//
#include <hip/hip_runtime.h>
#include <math.h>

typedef __attribute__((ext_vector_type(16))) _Float16 v16h;
typedef __attribute__((ext_vector_type(8)))  float    v8f;

#define B_  32
#define N_  256
#define D_  3
#define H_  64
#define EPS_ 1e-6f

__device__ __forceinline__ float fast_tanh(float x) {
#if __has_builtin(__builtin_amdgcn_tanhf)
  return __builtin_amdgcn_tanhf(x);
#elif __has_builtin(__builtin_amdgcn_tanh_f32)
  return __builtin_amdgcn_tanh_f32(x);
#else
  return tanhf(x);
#endif
}

__device__ __forceinline__ float fast_rsqrt(float x) {
#if __has_builtin(__builtin_amdgcn_rsqf)
  return __builtin_amdgcn_rsqf(x);
#else
  return rsqrtf(x);
#endif
}

// Fold-reduce 8 per-lane partials (v[g] = partial for pair g of this 16-lane
// half) summed over the 16 lanes of the half. After each xor step the data is
// invariant under that lane bit, so the bit is recycled to index a pair bit
// (one v_cndmask per pack). Returns: lane l holds full sum for pair (l&7)
// (pairs of this half). 15 shuffles + 7 selects vs 32 + 7 for butterfly+sel.
__device__ __forceinline__ float fold_reduce16(v8f v, int lane) {
  float a0 = v[0] + __shfl_xor(v[0], 1, 32);
  float a1 = v[1] + __shfl_xor(v[1], 1, 32);
  float a2 = v[2] + __shfl_xor(v[2], 1, 32);
  float a3 = v[3] + __shfl_xor(v[3], 1, 32);
  float a4 = v[4] + __shfl_xor(v[4], 1, 32);
  float a5 = v[5] + __shfl_xor(v[5], 1, 32);
  float a6 = v[6] + __shfl_xor(v[6], 1, 32);
  float a7 = v[7] + __shfl_xor(v[7], 1, 32);
  // pack pair-bit0 <- lane-bit0
  float b0 = (lane & 1) ? a1 : a0;
  float b1 = (lane & 1) ? a3 : a2;
  float b2 = (lane & 1) ? a5 : a4;
  float b3 = (lane & 1) ? a7 : a6;
  b0 += __shfl_xor(b0, 2, 32);
  b1 += __shfl_xor(b1, 2, 32);
  b2 += __shfl_xor(b2, 2, 32);
  b3 += __shfl_xor(b3, 2, 32);
  // pack pair-bit1 <- lane-bit1
  float c0 = (lane & 2) ? b1 : b0;
  float c1 = (lane & 2) ? b3 : b2;
  c0 += __shfl_xor(c0, 4, 32);
  c1 += __shfl_xor(c1, 4, 32);
  // pack pair-bit2 <- lane-bit2
  float d = (lane & 4) ? c1 : c0;
  d += __shfl_xor(d, 8, 32);
  return d;
}

// One wave (32 threads) per (b,i). Computes
//   v_ws[b,i,:]  = (1/(N-1)) * sum_j r_ij * f(d_ij)
//   tr_ws[b,i]   = (1/(N-1)) * sum_j (3 f + (|r|^2/d) f')   (includes j==i diagonal 3*f0)
__global__ __launch_bounds__(32) void DistanceField_field_kernel(
    const float* __restrict__ tptr, const float* __restrict__ x,
    const float* __restrict__ W1,   const float* __restrict__ b1,
    const float* __restrict__ W2,   const float* __restrict__ b2,
    const float* __restrict__ W3,   const float* __restrict__ b3,
    float* __restrict__ v_ws, float* __restrict__ tr_ws) {
  const int lane = threadIdx.x;
  const int lm   = lane & 15;
  const int half = lane >> 4;
  const int bi   = blockIdx.x;       // b*N + i
  const int b    = bi >> 8;

  const float t = tptr[0];

  // Layer-1 constants in the WMMA A-operand layout:
  // element e of A op holds K = (e&7) + 16*(e>>3) + 8*half + 32*op
  float a1[2][16], c1[2][16];
#pragma unroll
  for (int op = 0; op < 2; ++op)
#pragma unroll
    for (int e = 0; e < 16; ++e) {
      int k = (e & 7) + 16 * (e >> 3) + 8 * half + 32 * op;
      a1[op][e] = W1[k];                      // dW1 row (feature 0 = d)
      c1[op][e] = fmaf(W1[H_ + k], t, b1[k]); // t*W1[1,k] + b1[k]
    }

  // W2 in the WMMA B-operand layout (f16): element e -> K = (e&7) + 16*(e>>3) + 8*half + 32*kh,
  // column N = lm + 16*tt
  v16h bw[4][2];
#pragma unroll
  for (int tt = 0; tt < 4; ++tt)
#pragma unroll
    for (int kh = 0; kh < 2; ++kh) {
      int n = lm + 16 * tt;
#pragma unroll
      for (int e = 0; e < 16; ++e) {
        int k = (e & 7) + 16 * (e >> 3) + 8 * half + 32 * kh;
        bw[tt][kh][e] = (_Float16)W2[k * H_ + n];
      }
    }

  float w3v[4], b2v[4];
#pragma unroll
  for (int tt = 0; tt < 4; ++tt) {
    w3v[tt] = W3[lm + 16 * tt];
    b2v[tt] = b2[lm + 16 * tt];
  }
  const float b3v = b3[0];

  const float xi0 = x[bi * 3 + 0];
  const float xi1 = x[bi * 3 + 1];
  const float xi2 = x[bi * 3 + 2];
  const float* xb = x + b * (N_ * 3);

  const bool ownhalf = (((lane >> 3) & 1) == half);

  float vacc0 = 0.f, vacc1 = 0.f, vacc2 = 0.f, tracc = 0.f;

  for (int jt = 0; jt < 16; ++jt) {
    const int j = jt * 16 + lm;                 // pair row M = lm (both halves)
    const float r0 = xi0 - xb[j * 3 + 0];
    const float r1 = xi1 - xb[j * 3 + 1];
    const float r2 = xi2 - xb[j * 3 + 2];
    const float rr = r0 * r0 + r1 * r1 + r2 * r2;
    const float s  = rr + EPS_;
    const float invd = fast_rsqrt(s);   // 1/d
    const float d    = s * invd;        // d = sqrt(rr + eps)

    // layer 1 (+ JVP) directly into A operands
    v16h ah[2], dah[2];
#pragma unroll
    for (int op = 0; op < 2; ++op)
#pragma unroll
      for (int e = 0; e < 16; ++e) {
        float h1 = fast_tanh(fmaf(d, a1[op][e], c1[op][e]));
        ah[op][e]  = (_Float16)h1;
        dah[op][e] = (_Float16)((1.f - h1 * h1) * a1[op][e]);
      }

    v8f facc, dfacc;
#pragma unroll
    for (int g = 0; g < 8; ++g) { facc[g] = 0.f; dfacc[g] = 0.f; }

#pragma unroll
    for (int tt = 0; tt < 4; ++tt) {
      v8f c, dc;
#pragma unroll
      for (int g = 0; g < 8; ++g) { c[g] = 0.f; dc[g] = 0.f; }
      // layer-2 forward: h2pre = h1 @ W2   (K = 64 -> 2 WMMAs)
      c  = __builtin_amdgcn_wmma_f32_16x16x32_f16(false, ah[0],  false, bw[tt][0], (short)0, c,  false, false);
      c  = __builtin_amdgcn_wmma_f32_16x16x32_f16(false, ah[1],  false, bw[tt][1], (short)0, c,  false, false);
      // layer-2 JVP: dh2pre = dh1 @ W2
      dc = __builtin_amdgcn_wmma_f32_16x16x32_f16(false, dah[0], false, bw[tt][0], (short)0, dc, false, false);
      dc = __builtin_amdgcn_wmma_f32_16x16x32_f16(false, dah[1], false, bw[tt][1], (short)0, dc, false, false);
      // layer 3 (+ JVP) on the C-layout tile: lane holds N = lm+16*tt for rows g(+8*half)
#pragma unroll
      for (int g = 0; g < 8; ++g) {
        float h2 = fast_tanh(c[g] + b2v[tt]);
        facc[g]  = fmaf(h2, w3v[tt], facc[g]);
        dfacc[g] = fmaf((1.f - h2 * h2) * dc[g], w3v[tt], dfacc[g]);
      }
    }

    // fold-reduce hidden-dim partials: lane ends holding pair (lane&7) of its
    // half; one xor-16 swap resolves lanes whose pair lives on the other half.
    float fs  = fold_reduce16(facc,  lane);
    float dfs = fold_reduce16(dfacc, lane);
    float fo  = __shfl_xor(fs,  16, 32);
    float dfo = __shfl_xor(dfs, 16, 32);
    const float fj  = (ownhalf ? fs : fo) + b3v;
    const float dfj =  ownhalf ? dfs : dfo;

    vacc0 = fmaf(r0, fj, vacc0);
    vacc1 = fmaf(r1, fj, vacc1);
    vacc2 = fmaf(r2, fj, vacc2);
    tracc += 3.f * fj + (rr * invd) * dfj;  // j==i: rr=0 -> adds 3*f0 (removed later)
  }

  // both halves hold identical partial sums; reduce within a half
#pragma unroll
  for (int m = 1; m < 16; m <<= 1) {
    vacc0 += __shfl_xor(vacc0, m, 32);
    vacc1 += __shfl_xor(vacc1, m, 32);
    vacc2 += __shfl_xor(vacc2, m, 32);
    tracc += __shfl_xor(tracc, m, 32);
  }
  if (lane == 0) {
    const float inv = 1.f / (float)(N_ - 1);
    v_ws[bi * 3 + 0] = vacc0 * inv;
    v_ws[bi * 3 + 1] = vacc1 * inv;
    v_ws[bi * 3 + 2] = vacc2 * inv;
    tr_ws[bi]        = tracc * inv;
  }
}

// One block per batch b: mean-removal, diagonal correction, trace assembly.
__global__ __launch_bounds__(256) void DistanceField_finalize_kernel(
    const float* __restrict__ tptr, const float* __restrict__ W1,
    const float* __restrict__ b1,   const float* __restrict__ W2,
    const float* __restrict__ b2,   const float* __restrict__ W3,
    const float* __restrict__ b3,   const float* __restrict__ v_ws,
    const float* __restrict__ tr_ws, float* __restrict__ out) {
  __shared__ float sh_h1[H_];
  __shared__ float sh_p[H_];
  __shared__ float sh_f0;
  __shared__ float sh_mean[3];
  __shared__ float sh_trsum;
  __shared__ float red[8][4];

  const int b   = blockIdx.x;
  const int tid = threadIdx.x;
  const float t  = tptr[0];
  const float d0 = sqrtf(EPS_);   // diagonal distance

  // f0 = MLP(d0, t), computed cooperatively by 64 threads
  if (tid < H_) sh_h1[tid] = fast_tanh(fmaf(W1[tid], d0, fmaf(W1[H_ + tid], t, b1[tid])));
  __syncthreads();
  if (tid < H_) {
    float acc = b2[tid];
#pragma unroll 8
    for (int k = 0; k < H_; ++k) acc = fmaf(sh_h1[k], W2[k * H_ + tid], acc);
    sh_p[tid] = fast_tanh(acc) * W3[tid];
  }
  __syncthreads();
  if (tid == 0) {
    float s = b3[0];
    for (int k = 0; k < H_; ++k) s += sh_p[k];
    sh_f0 = s;
  }

  // per-batch reductions of v and tr (tid == particle i)
  const float v0 = v_ws[(b * N_ + tid) * 3 + 0];
  const float v1 = v_ws[(b * N_ + tid) * 3 + 1];
  const float v2 = v_ws[(b * N_ + tid) * 3 + 2];
  const float tr = tr_ws[b * N_ + tid];

  float s0 = v0, s1 = v1, s2 = v2, s3 = tr;
#pragma unroll
  for (int m = 1; m < 32; m <<= 1) {
    s0 += __shfl_xor(s0, m, 32);
    s1 += __shfl_xor(s1, m, 32);
    s2 += __shfl_xor(s2, m, 32);
    s3 += __shfl_xor(s3, m, 32);
  }
  if ((tid & 31) == 0) {
    red[tid >> 5][0] = s0; red[tid >> 5][1] = s1;
    red[tid >> 5][2] = s2; red[tid >> 5][3] = s3;
  }
  __syncthreads();
  if (tid == 0) {
    float m0 = 0.f, m1 = 0.f, m2 = 0.f, m3 = 0.f;
#pragma unroll
    for (int w = 0; w < 8; ++w) { m0 += red[w][0]; m1 += red[w][1]; m2 += red[w][2]; m3 += red[w][3]; }
    sh_mean[0] = m0 / (float)N_;
    sh_mean[1] = m1 / (float)N_;
    sh_mean[2] = m2 / (float)N_;
    sh_trsum   = m3;
  }
  __syncthreads();

  // y = v - mean_i(v)
  out[(b * N_ + tid) * 3 + 0] = v0 - sh_mean[0];
  out[(b * N_ + tid) * 3 + 1] = v1 - sh_mean[1];
  out[(b * N_ + tid) * 3 + 2] = v2 - sh_mean[2];
  if (tid == 0) {
    const float diag = (float)N_ * 3.f * sh_f0 / (float)(N_ - 1);
    out[B_ * N_ * 3 + b] = (sh_trsum - diag) * (1.f - 1.f / (float)N_);
  }
}

extern "C" void kernel_launch(void* const* d_in, const int* in_sizes, int n_in,
                              void* d_out, int out_size, void* d_ws, size_t ws_size,
                              hipStream_t stream) {
  (void)in_sizes; (void)n_in; (void)out_size; (void)ws_size;
  const float* t  = (const float*)d_in[0];
  const float* x  = (const float*)d_in[1];
  const float* W1 = (const float*)d_in[2];
  const float* b1 = (const float*)d_in[3];
  const float* W2 = (const float*)d_in[4];
  const float* b2 = (const float*)d_in[5];
  const float* W3 = (const float*)d_in[6];
  const float* b3 = (const float*)d_in[7];

  float* v_ws  = (float*)d_ws;            // [B, N, 3]
  float* tr_ws = v_ws + B_ * N_ * 3;      // [B, N]

  DistanceField_field_kernel<<<B_ * N_, 32, 0, stream>>>(
      t, x, W1, b1, W2, b2, W3, b3, v_ws, tr_ws);
  DistanceField_finalize_kernel<<<B_, 256, 0, stream>>>(
      t, W1, b1, W2, b2, W3, b3, v_ws, tr_ws, (float*)d_out);
}
